// GraphRNNBilinear_16844861735658
// MI455X (gfx1250) — matrix operand order repaired and checked by reading, not compiled, not run on hardware
//
#include <hip/hip_runtime.h>
#include <hip/hip_bf16.h>
#include <math.h>

// ---------------------------------------------------------------------------
// Problem constants
// ---------------------------------------------------------------------------
#define BB   16
#define S0v  511
#define SS   512
#define DD   512
#define HGv  256
#define HEv  256
#define TT   128
#define MM   32
#define NN   (BB * SS)          // 8192
#define AST  (MM + 1)           // 33, A row stride

typedef __attribute__((ext_vector_type(16))) __bf16 v16bf;
typedef __attribute__((ext_vector_type(8)))  __bf16 v8bf;
typedef __attribute__((ext_vector_type(8)))  float  v8f;

// ---------------------------------------------------------------------------
// WMMA helpers (CDNA5 wave32, 16x16x32 bf16 -> f32)
// A fragment as these helpers load it (contiguous k, not the ISA table order) (16x32, MxK): lane<16: row=lane, elems = K0..15;
// lane>=16: row=lane-16, elems = K16..31.
// B fragment as loaded here, same k order as A (32x16, KxN): lane<16: col=lane, elems = K0..15;
// lane>=16: col=lane-16, elems = K16..31.  B[k][n] = W[n][k] (row-major W).
// C/D layout: VGPR v -> row = (lane>>4)*8 + v, col = lane&15.
// ---------------------------------------------------------------------------
__device__ __forceinline__ v8f wmma_bf16(v16bf a, v16bf b, v8f c) {
  return __builtin_amdgcn_wmma_f32_16x16x32_bf16(
      /*neg_a=*/false, a, /*neg_b=*/false, b,
      /*c_mod=*/(short)0, c, /*reuse_a=*/false, /*reuse_b=*/false);
}

__device__ __forceinline__ v16bf load_a_frag(const __bf16* baseRow0, int ld,
                                             int k0, int lane) {
  int r    = lane & 15;
  int half = lane >> 4;
  const __bf16* p = baseRow0 + (size_t)r * ld + k0 + half * 16;
  v8bf lo = *reinterpret_cast<const v8bf*>(p);
  v8bf hi = *reinterpret_cast<const v8bf*>(p + 8);
  return __builtin_shufflevector(lo, hi, 0, 1, 2, 3, 4, 5, 6, 7,
                                         8, 9, 10, 11, 12, 13, 14, 15);
}

__device__ __forceinline__ v16bf load_b_frag(const __bf16* W, int ld,
                                             int n0, int k0, int lane) {
  int col  = n0 + (lane & 15);
  int half = lane >> 4;
  const __bf16* p = W + (size_t)col * ld + k0 + half * 16;
  return *reinterpret_cast<const v16bf*>(p);
}

// ---------------------------------------------------------------------------
// Generic fused WMMA GEMM:
//   Out[N x O] = act( X1[NxK1] @ W1[OxK1]^T  (+ X2[NxK2] @ W2[OxK2]^T)
//                     + bias1 + bias2 + pvec[row]*uvec[col] )
// Each wave computes a 16x64 tile (4 accumulators). Block = 8 waves.
// act: 0 = none, 1 = tanh, 2 = elu
// ---------------------------------------------------------------------------
__global__ void __launch_bounds__(256)
wmma_gemm(const __bf16* __restrict__ X1, const __bf16* __restrict__ W1, int K1,
          const __bf16* __restrict__ X2, const __bf16* __restrict__ W2, int K2,
          const float* __restrict__ bias1, const float* __restrict__ bias2,
          const float* __restrict__ pvec,  const float* __restrict__ uvec,
          float* __restrict__ outF, __bf16* __restrict__ outB,
          int O, int act, int totalTiles) {
  int wave = threadIdx.x >> 5;
  int lane = threadIdx.x & 31;
  int tile = blockIdx.x * 8 + wave;
  if (tile >= totalTiles) return;                 // wave-uniform

  int colGroups = O >> 6;
  int rowT = tile / colGroups;
  int colG = tile - rowT * colGroups;
  int m0 = rowT << 4;
  int n0 = colG << 6;

  v8f acc[4] = {};

  const __bf16* Xb = X1 + (size_t)m0 * K1;
  for (int k0 = 0; k0 < K1; k0 += 32) {
    v16bf a = load_a_frag(Xb, K1, k0, lane);
#pragma unroll
    for (int j = 0; j < 4; ++j) {
      v16bf b = load_b_frag(W1, K1, n0 + j * 16, k0, lane);
      acc[j] = wmma_bf16(a, b, acc[j]);
    }
  }
  if (X2) {
    const __bf16* Xb2 = X2 + (size_t)m0 * K2;
    for (int k0 = 0; k0 < K2; k0 += 32) {
      v16bf a = load_a_frag(Xb2, K2, k0, lane);
#pragma unroll
      for (int j = 0; j < 4; ++j) {
        v16bf b = load_b_frag(W2, K2, n0 + j * 16, k0, lane);
        acc[j] = wmma_bf16(a, b, acc[j]);
      }
    }
  }

  int cIn   = lane & 15;
  int rHalf = lane >> 4;
#pragma unroll
  for (int j = 0; j < 4; ++j) {
    int col = n0 + j * 16 + cIn;
    float badd = 0.f;
    if (bias1) badd += bias1[col];
    if (bias2) badd += bias2[col];
    float u = uvec ? uvec[col] : 0.f;
#pragma unroll
    for (int v = 0; v < 8; ++v) {
      int row = m0 + rHalf * 8 + v;
      float val = acc[j][v] + badd;
      if (pvec) val += pvec[row] * u;
      if (act == 1)      val = tanhf(val);
      else if (act == 2) val = (val > 0.f) ? val : expm1f(val);
      size_t o = (size_t)row * O + col;
      if (outF) outF[o] = val;
      if (outB) outB[o] = (__bf16)val;
    }
  }
}

// ---------------------------------------------------------------------------
// Persistent single-workgroup RNN scan (one layer):
//   h_t = tanh(xw[:, t, :] + h_{t-1} @ Whh^T)   for t = 0..511, h in R^{16x256}
// Whh (bf16) + h state live in LDS (139 KB < 320 KB WGP budget).
// 8 waves; wave w owns output columns [w*32, w*32+32).
// Output gs (bf16), optionally masked (layer-1 only).
// ---------------------------------------------------------------------------
__global__ void __launch_bounds__(256)
rnn_scan(const float* __restrict__ xw, const float* __restrict__ Whh,
         __bf16* __restrict__ outB, const int* __restrict__ mask, int useMask) {
  __shared__ __bf16 sW[HGv * HGv];   // 128 KB
  __shared__ __bf16 sh[BB * HGv];    // 8 KB
  int tid = threadIdx.x;
  for (int i = tid; i < HGv * HGv; i += 256) sW[i] = (__bf16)Whh[i];
  for (int i = tid; i < BB * HGv; i += 256)  sh[i] = (__bf16)0.f;
  __syncthreads();

  int wave  = tid >> 5;
  int lane  = tid & 31;
  int cIn   = lane & 15;
  int rHalf = lane >> 4;
  int n0    = wave * 32;

  for (int t = 0; t < SS; ++t) {
    v8f acc[2];
#pragma unroll
    for (int j = 0; j < 2; ++j) {
      int col = n0 + j * 16 + cIn;
#pragma unroll
      for (int v = 0; v < 8; ++v) {
        int b = rHalf * 8 + v;
        acc[j][v] = xw[((size_t)(b * SS + t)) * HGv + col];
      }
    }
    for (int k0 = 0; k0 < HGv; k0 += 32) {
      v16bf a = load_a_frag(sh, HGv, k0, lane);
#pragma unroll
      for (int j = 0; j < 2; ++j) {
        v16bf b = load_b_frag(sW, HGv, n0 + j * 16, k0, lane);
        acc[j] = wmma_bf16(a, b, acc[j]);
      }
    }
    float hv[2][8];
#pragma unroll
    for (int j = 0; j < 2; ++j)
#pragma unroll
      for (int v = 0; v < 8; ++v) hv[j][v] = tanhf(acc[j][v]);

    __syncthreads();  // all waves done reading h_{t-1}
#pragma unroll
    for (int j = 0; j < 2; ++j) {
      int col = n0 + j * 16 + cIn;
#pragma unroll
      for (int v = 0; v < 8; ++v) {
        int b = rHalf * 8 + v;
        sh[b * HGv + col] = (__bf16)hv[j][v];
        float ov = hv[j][v];
        if (useMask) {
          float mf = (t == 0) ? 1.f : (float)mask[b * S0v + (t - 1)];
          ov *= mf;
        }
        outB[((size_t)(b * SS + t)) * HGv + col] = (__bf16)ov;
      }
    }
    __syncthreads();  // new h visible before next step
  }
}

// ---------------------------------------------------------------------------
// Build x = concat(sentinel, input) in bf16.  n = b*S + t.
// ---------------------------------------------------------------------------
__global__ void __launch_bounds__(256)
build_x(const float* __restrict__ input, const float* __restrict__ sentinel,
        __bf16* __restrict__ xbf) {
  size_t idx = (size_t)blockIdx.x * 256 + threadIdx.x;
  if (idx >= (size_t)NN * DD) return;
  int d = (int)(idx % DD);
  int n = (int)(idx / DD);
  int b = n / SS, t = n % SS;
  float v = (t == 0) ? sentinel[d] : input[((size_t)(b * S0v + (t - 1))) * DD + d];
  xbf[idx] = (__bf16)v;
}

__global__ void __launch_bounds__(256)
cvt_f32_bf16(const float* __restrict__ src, __bf16* __restrict__ dst, int n) {
  int i = blockIdx.x * 256 + threadIdx.x;
  if (i < n) dst[i] = (__bf16)src[i];
}

// sigmoid classifier: p[n] = sigmoid(h1[n,:] . clsW + clsb); one wave per row
__global__ void __launch_bounds__(256)
cls_prob(const __bf16* __restrict__ h1, const float* __restrict__ clsW,
         const float* __restrict__ clsb, float* __restrict__ pout,
         float* __restrict__ Abuf, int mcol) {
  int wave = threadIdx.x >> 5;
  int lane = threadIdx.x & 31;
  int row  = blockIdx.x * 8 + wave;
  const __bf16* hr = h1 + (size_t)row * HEv;
  float s = 0.f;
  for (int i = lane; i < HEv; i += 32) s += (float)hr[i] * clsW[i];
  for (int off = 16; off; off >>= 1) s += __shfl_xor(s, off, 32);
  if (lane == 0) {
    float p = 1.f / (1.f + expf(-(s + clsb[0])));
    pout[row] = p;
    Abuf[(size_t)row * AST + mcol] = p;
  }
}

__global__ void __launch_bounds__(256)
estep_init(float* __restrict__ Abuf, float* __restrict__ p0,
           __bf16* __restrict__ h1z) {
  int idx = blockIdx.x * 256 + threadIdx.x;   // 8192 threads
  if (idx < NN) { Abuf[(size_t)idx * AST] = 1.f; p0[idx] = 1.f; }
  for (size_t i = idx; i < (size_t)NN * HEv; i += (size_t)gridDim.x * 256)
    h1z[i] = (__bf16)0.f;
}

// arc_logits[b, j, i] = (j >= max(0,i-M) && j < i) ? A[b*S+i][(j-start)+1] : 0
__global__ void __launch_bounds__(256)
arc_gather(const float* __restrict__ Abuf, float* __restrict__ arc) {
  size_t idx = (size_t)blockIdx.x * 256 + threadIdx.x;
  if (idx >= (size_t)BB * SS * SS) return;
  int i = (int)(idx % SS);
  int j = (int)((idx / SS) % SS);
  int b = (int)(idx / ((size_t)SS * SS));
  int start = i - MM; if (start < 0) start = 0;
  float val = 0.f;
  if (j >= start && j < i) {
    int m = j - start; if (m > MM - 1) m = MM - 1;
    val = Abuf[((size_t)(b * SS + i)) * AST + m + 1];
  }
  arc[idx] = val;
}

// ---------------------------------------------------------------------------
// Host launcher
// ---------------------------------------------------------------------------
extern "C" void kernel_launch(void* const* d_in, const int* in_sizes, int n_in,
                              void* d_out, int out_size, void* d_ws, size_t ws_size,
                              hipStream_t stream) {
  (void)in_sizes; (void)n_in; (void)out_size; (void)ws_size;
  const float* input    = (const float*)d_in[0];
  const float* sentinel = (const float*)d_in[1];
  const float* gWih0    = (const float*)d_in[2];
  const float* gWhh0    = (const float*)d_in[3];
  const float* gbih0    = (const float*)d_in[4];
  const float* gbhh0    = (const float*)d_in[5];
  const float* gWih1    = (const float*)d_in[6];
  const float* gWhh1    = (const float*)d_in[7];
  const float* gbih1    = (const float*)d_in[8];
  const float* gbhh1    = (const float*)d_in[9];
  const float* eWih0    = (const float*)d_in[10];   // (HE,1) -> rank-1 u
  const float* eWhh0    = (const float*)d_in[11];
  const float* ebih0    = (const float*)d_in[12];
  const float* ebhh0    = (const float*)d_in[13];
  const float* eWih1    = (const float*)d_in[14];
  const float* eWhh1    = (const float*)d_in[15];
  const float* ebih1    = (const float*)d_in[16];
  const float* ebhh1    = (const float*)d_in[17];
  const float* clsW     = (const float*)d_in[18];
  const float* clsb     = (const float*)d_in[19];
  const float* headW    = (const float*)d_in[20];
  const float* headb    = (const float*)d_in[21];
  const float* depW     = (const float*)d_in[22];
  const float* depb     = (const float*)d_in[23];
  const int*   mask     = (const int*)d_in[24];

  float* outHead = (float*)d_out;                          // (B,S,T)
  float* outDep  = outHead + (size_t)BB * SS * TT;         // (B,S,T)
  float* outArc  = outDep  + (size_t)BB * SS * TT;         // (B,S,S)

  // ---- workspace bump allocator -----------------------------------------
  char* wsp = (char*)d_ws;
  auto alloc = [&](size_t bytes) -> char* {
    char* p = wsp;
    wsp += (bytes + 255) & ~(size_t)255;
    return p;
  };
  __bf16* xbf      = (__bf16*)alloc((size_t)NN * DD * 2);
  __bf16* h0seq    = (__bf16*)alloc((size_t)NN * HGv * 2);
  __bf16* gsbf     = (__bf16*)alloc((size_t)NN * HGv * 2);
  float*  xw0      = (float*) alloc((size_t)NN * HGv * 4);
  float*  xw1      = (float*) alloc((size_t)NN * HGv * 4);
  __bf16* h0b[2]   = { (__bf16*)alloc((size_t)NN * HEv * 2),
                       (__bf16*)alloc((size_t)NN * HEv * 2) };
  __bf16* h1b[2]   = { (__bf16*)alloc((size_t)NN * HEv * 2),
                       (__bf16*)alloc((size_t)NN * HEv * 2) };
  float*  pb[2]    = { (float*)alloc((size_t)NN * 4), (float*)alloc((size_t)NN * 4) };
  float*  Abuf     = (float*) alloc((size_t)NN * AST * 4);
  __bf16* bWih0    = (__bf16*)alloc((size_t)HGv * DD  * 2);
  __bf16* bWhh0    = (__bf16*)alloc((size_t)HGv * HGv * 2);  (void)bWhh0;
  __bf16* bWih1    = (__bf16*)alloc((size_t)HGv * HGv * 2);
  __bf16* beWhh0   = (__bf16*)alloc((size_t)HEv * HEv * 2);
  __bf16* beWih1   = (__bf16*)alloc((size_t)HEv * HEv * 2);
  __bf16* beWhh1   = (__bf16*)alloc((size_t)HEv * HEv * 2);
  __bf16* bheadW   = (__bf16*)alloc((size_t)TT * HGv * 2);
  __bf16* bdepW    = (__bf16*)alloc((size_t)TT * HGv * 2);

  auto cvt = [&](const float* s, __bf16* d, int n) {
    cvt_f32_bf16<<<(n + 255) / 256, 256, 0, stream>>>(s, d, n);
  };
  cvt(gWih0, bWih0, HGv * DD);
  cvt(gWih1, bWih1, HGv * HGv);
  cvt(eWhh0, beWhh0, HEv * HEv);
  cvt(eWih1, beWih1, HEv * HEv);
  cvt(eWhh1, beWhh1, HEv * HEv);
  cvt(headW, bheadW, TT * HGv);
  cvt(depW,  bdepW,  TT * HGv);

  // ---- build x (sentinel prepended), bf16 -------------------------------
  build_x<<<((size_t)NN * DD + 255) / 256, 256, 0, stream>>>(input, sentinel, xbf);

  // ---- xw0 = x @ Wih0^T + biases  (8192x512x256) ------------------------
  {
    int tiles = (NN / 16) * (HGv / 64);    // 2048
    wmma_gemm<<<tiles / 8, 256, 0, stream>>>(
        xbf, bWih0, DD, nullptr, nullptr, 0, gbih0, gbhh0,
        nullptr, nullptr, xw0, nullptr, HGv, 0, tiles);
  }
  // ---- layer-0 scan (persistent WG, Whh0 f32 source converted in-LDS) ----
  rnn_scan<<<1, 256, 0, stream>>>(xw0, gWhh0, h0seq, nullptr, 0);

  // ---- xw1 = h0seq @ Wih1^T + biases ------------------------------------
  {
    int tiles = (NN / 16) * (HGv / 64);
    wmma_gemm<<<tiles / 8, 256, 0, stream>>>(
        h0seq, bWih1, HGv, nullptr, nullptr, 0, gbih1, gbhh1,
        nullptr, nullptr, xw1, nullptr, HGv, 0, tiles);
  }
  // ---- layer-1 scan, masked output = gs ---------------------------------
  rnn_scan<<<1, 256, 0, stream>>>(xw1, gWhh1, gsbf, mask, 1);

  // ---- estep: 32 sequential (GEMM, GEMM, sigmoid) rounds ----------------
  estep_init<<<32, 256, 0, stream>>>(Abuf, pb[1], h1b[1]);
  int tilesE = (NN / 16) * (HEv / 64);     // 2048
  for (int m = 0; m < MM; ++m) {
    const __bf16* h0in = (m == 0) ? gsbf : h0b[(m + 1) & 1];
    const __bf16* h1in = h1b[(m + 1) & 1];
    const float*  pin  = pb[(m + 1) & 1];
    __bf16* h0out = h0b[m & 1];
    __bf16* h1out = h1b[m & 1];
    float*  pout  = pb[m & 1];
    // h0' = tanh(h0 @ eWhh0^T + bih0 + bhh0 + p * eWih0[:,0])
    wmma_gemm<<<tilesE / 8, 256, 0, stream>>>(
        h0in, beWhh0, HEv, nullptr, nullptr, 0, ebih0, ebhh0,
        pin, eWih0, nullptr, h0out, HEv, 1, tilesE);
    // h1' = tanh(h0' @ eWih1^T + h1 @ eWhh1^T + bih1 + bhh1)
    wmma_gemm<<<tilesE / 8, 256, 0, stream>>>(
        h0out, beWih1, HEv, h1in, beWhh1, HEv, ebih1, ebhh1,
        nullptr, nullptr, nullptr, h1out, HEv, 1, tilesE);
    // p' = sigmoid(h1' . clsW + clsb) ; also fills A column m+1
    cls_prob<<<NN / 8, 256, 0, stream>>>(h1out, clsW, clsb, pout, Abuf, m + 1);
  }

  // ---- head/dep tags: elu(gs @ W^T + b) ---------------------------------
  {
    int tiles = (NN / 16) * (TT / 64);     // 1024
    wmma_gemm<<<tiles / 8, 256, 0, stream>>>(
        gsbf, bheadW, HGv, nullptr, nullptr, 0, headb, nullptr,
        nullptr, nullptr, outHead, nullptr, TT, 2, tiles);
    wmma_gemm<<<tiles / 8, 256, 0, stream>>>(
        gsbf, bdepW, HGv, nullptr, nullptr, 0, depb, nullptr,
        nullptr, nullptr, outDep, nullptr, TT, 2, tiles);
  }

  // ---- arc_logits gather ------------------------------------------------
  arc_gather<<<((size_t)BB * SS * SS + 255) / 256, 256, 0, stream>>>(Abuf, outArc);
}
